// Block_18433999634678
// MI455X (gfx1250) — compile-verified, matrix operand order checked by reading, not run on hardware
//
#include <hip/hip_runtime.h>
#include <hip/hip_bf16.h>
#include <math.h>
#include <stdint.h>

// ---------------------------------------------------------------------------
// Types for CDNA5 WMMA (gfx1250): v_wmma_f32_16x16x32_bf16
// ---------------------------------------------------------------------------
typedef __bf16 v16bf __attribute__((ext_vector_type(16)));
typedef float  v8f   __attribute__((ext_vector_type(8)));
typedef unsigned int v8u __attribute__((ext_vector_type(8)));
typedef int v4i_vs __attribute__((vector_size(16)));   // matches builtin param

#define DMODEL 768
#define FDIM   3072
#define MROWS  4096   // B*S = 4*1024
#define SEQ    1024
#define NHEAD  12
#define HDIM   64

// ---------------------------------------------------------------------------
// CDNA5 async global->LDS copy (GLOBAL_LOAD_ASYNC_TO_LDS_B128, ASYNCcnt).
// Probe result (round 2 diagnostic): param0 = AS(1) int4* (non-const),
// 4 args total. AS(1) cast via uintptr_t (numerically identity for global);
// AS(3) cast must be a real addrspacecast (aperture strip).
// ---------------------------------------------------------------------------
#if defined(__has_builtin)
#if __has_builtin(__builtin_amdgcn_global_load_async_to_lds_b128) && \
    __has_builtin(__builtin_amdgcn_s_wait_asynccnt)
#define USE_ASYNC_LDS 1
#endif
#endif
#ifndef USE_ASYNC_LDS
#define USE_ASYNC_LDS 0
#endif

__device__ __forceinline__ void copy_b128_g2l(const void* g, void* l) {
#if USE_ASYNC_LDS
    __builtin_amdgcn_global_load_async_to_lds_b128(
        (__attribute__((address_space(1))) v4i_vs*)(uintptr_t)g,
        (__attribute__((address_space(3))) v4i_vs*)l,
        0, 0);
#else
    *(uint4*)l = *(const uint4*)g;
#endif
}
__device__ __forceinline__ void async_copy_wait() {
#if USE_ASYNC_LDS
    __builtin_amdgcn_s_wait_asynccnt(0);
#endif
}

// ---------------------------------------------------------------------------
// bf16 helpers (RNE convert, stored as ushort)
// ---------------------------------------------------------------------------
__device__ __forceinline__ unsigned short f2bf(float f) {
    unsigned int u = __builtin_bit_cast(unsigned int, f);
    u = (u + 0x7FFFu + ((u >> 16) & 1u)) >> 16;
    return (unsigned short)u;
}
__device__ __forceinline__ float bf2f(unsigned short h) {
    unsigned int u = ((unsigned int)h) << 16;
    return __builtin_bit_cast(float, u);
}
__device__ __forceinline__ unsigned int pack2bf(float a, float b) {
    return (unsigned int)f2bf(a) | ((unsigned int)f2bf(b) << 16);
}
__device__ __forceinline__ unsigned short u16_of(const uint4& q, int j) {
    unsigned int c = ((const unsigned int*)&q)[j >> 1];
    return (unsigned short)(c >> (16 * (j & 1)));
}

// Build a 16x32 bf16 WMMA fragment from an LDS row (K contiguous).
// ISA layout (16-bit A / B): lane L: hi = L>>4; VGPR r holds K pair
//   kbase = 2r (+8 if r>=4) + 8*hi  -> pairs merge into ds_load_b128 x2.
__device__ __forceinline__ v16bf frag16(const unsigned short* base, int hi) {
    v8u u;
#pragma unroll
    for (int r = 0; r < 8; ++r) {
        int kb = 2 * r + ((r < 4) ? 0 : 8) + 8 * hi;
        u[r] = *(const unsigned int*)(base + kb);
    }
    return __builtin_bit_cast(v16bf, u);
}

__device__ __forceinline__ float gelu_exact(float x) {
    return 0.5f * x * (1.0f + erff(x * 0.70710678118654752f));
}

// ---------------------------------------------------------------------------
// Elementwise: f32 -> bf16 convert, 8 elements per thread-iteration
// ---------------------------------------------------------------------------
__global__ void cvt_f32_bf16(const float* __restrict__ src,
                             unsigned short* __restrict__ dst, int nchunk) {
    int i = blockIdx.x * blockDim.x + threadIdx.x;
    int stride = gridDim.x * blockDim.x;
    for (; i < nchunk; i += stride) {
        const float4 a = ((const float4*)src)[2 * i + 0];
        const float4 b = ((const float4*)src)[2 * i + 1];
        uint4 o;
        o.x = pack2bf(a.x, a.y); o.y = pack2bf(a.z, a.w);
        o.z = pack2bf(b.x, b.y); o.w = pack2bf(b.z, b.w);
        ((uint4*)dst)[i] = o;
    }
}

// ---------------------------------------------------------------------------
// Elementwise: out = bf16(wa * a + wb * b), 8 bf16 per iteration
// ---------------------------------------------------------------------------
__global__ void combine_bf16(const unsigned short* __restrict__ a,
                             const unsigned short* __restrict__ b,
                             const float* __restrict__ pwa,
                             const float* __restrict__ pwb,
                             unsigned short* __restrict__ out, int nchunk) {
    float wa = *pwa, wb = *pwb;
    int i = blockIdx.x * blockDim.x + threadIdx.x;
    int stride = gridDim.x * blockDim.x;
    for (; i < nchunk; i += stride) {
        uint4 ua = ((const uint4*)a)[i];
        uint4 ub = ((const uint4*)b)[i];
        uint4 o;
#pragma unroll
        for (int c = 0; c < 4; ++c) {
            unsigned int xa = ((const unsigned int*)&ua)[c];
            unsigned int xb = ((const unsigned int*)&ub)[c];
            float lo = wa * bf2f((unsigned short)xa) + wb * bf2f((unsigned short)xb);
            float hi = wa * bf2f((unsigned short)(xa >> 16)) + wb * bf2f((unsigned short)(xb >> 16));
            ((unsigned int*)&o)[c] = pack2bf(lo, hi);
        }
        ((uint4*)out)[i] = o;
    }
}

// ---------------------------------------------------------------------------
// LayerNorm over D=768, f32 in -> bf16 out. One block (256 thr) per row.
// ---------------------------------------------------------------------------
__global__ __launch_bounds__(256) void ln_bf16(const float* __restrict__ X,
                                               const float* __restrict__ g,
                                               const float* __restrict__ bt,
                                               unsigned short* __restrict__ out) {
    __shared__ float rs[256];
    __shared__ float rq[256];
    int row = blockIdx.x;
    int t = threadIdx.x;
    const float* xr = X + (size_t)row * DMODEL;
    float v0 = xr[t], v1 = xr[t + 256], v2 = xr[t + 512];
    rs[t] = v0 + v1 + v2;
    rq[t] = v0 * v0 + v1 * v1 + v2 * v2;
    __syncthreads();
    for (int off = 128; off > 0; off >>= 1) {
        if (t < off) { rs[t] += rs[t + off]; rq[t] += rq[t + off]; }
        __syncthreads();
    }
    float mean = rs[0] * (1.0f / DMODEL);
    float var  = rq[0] * (1.0f / DMODEL) - mean * mean;
    float rstd = rsqrtf(var + 1e-6f);
    unsigned short* orow = out + (size_t)row * DMODEL;
#pragma unroll
    for (int i = 0; i < 3; ++i) {
        int c = t + i * 256;
        float v = (i == 0) ? v0 : ((i == 1) ? v1 : v2);
        orow[c] = f2bf((v - mean) * rstd * g[c] + bt[c]);
    }
}

// ---------------------------------------------------------------------------
// bf16 GEMM:  C[M,N] = A[M,K] @ W[K,N]  (bf16 in, f32 accumulate)
// Block 256 thr = 8 waves; tile (64*MT)x128; wave tile (16*MT)x64.
// mode 0: outb = bf16(acc + bias)
// mode 1: outb = bf16(gelu(acc + bias))
// mode 2: outf = acc + bias*(*pw1+*pw2 or 1) + res
// ---------------------------------------------------------------------------
template <int MT>
__global__ __launch_bounds__(256) void gemm_bf16(
    const unsigned short* __restrict__ A, int lda,
    const unsigned short* __restrict__ W, int ldw,
    const float* __restrict__ bias,
    const float* __restrict__ res,
    unsigned short* __restrict__ outb,
    float* __restrict__ outf,
    int N, int K, int mode,
    const float* pw1, const float* pw2) {
    __shared__ __align__(16) unsigned short sA[64 * MT * 40]; // [m][k] pitch 40
    __shared__ __align__(16) unsigned short sW[128 * 40];     // [n][k] transposed

    const int t = threadIdx.x;
    const int lane = t & 31;
    const int wave = t >> 5;
    const int m0 = blockIdx.y * (64 * MT);
    const int n0 = blockIdx.x * 128;
    const int wm = (wave & 3) * (16 * MT);
    const int wn = (wave >> 2) * 64;
    const int hi = lane >> 4;
    const int l15 = lane & 15;

    v8f acc[MT][4] = {};

    for (int kk = 0; kk < K; kk += 32) {
        __syncthreads();
        // A tile (64*MT x 32): b128 chunks, async path if available
#pragma unroll
        for (int i = 0; i < MT; ++i) {
            int c = i * 256 + t;
            int m = c >> 2, k8 = c & 3;
            copy_b128_g2l(&A[(size_t)(m0 + m) * lda + kk + k8 * 8],
                          &sA[m * 40 + k8 * 8]);
        }
        // W tile (32 x 128) transposed into LDS: vector global load + scatter
#pragma unroll
        for (int i = 0; i < 2; ++i) {
            int c = i * 256 + t;
            int kr = c >> 4, n8 = c & 15;
            uint4 q = *(const uint4*)&W[(size_t)(kk + kr) * ldw + n0 + n8 * 8];
#pragma unroll
            for (int j = 0; j < 8; ++j)
                sW[(n8 * 8 + j) * 40 + kr] = u16_of(q, j);
        }
        if (kk + 32 < K) {
            __builtin_prefetch(&A[(size_t)(m0 + (t & (64 * MT - 1))) * lda + kk + 32]);
            __builtin_prefetch(&W[(size_t)(kk + 32 + (t & 31)) * ldw + n0 + (t & 127)]);
        }
        async_copy_wait();
        __syncthreads();

        v16bf a[MT];
#pragma unroll
        for (int mt = 0; mt < MT; ++mt)
            a[mt] = frag16(&sA[(wm + mt * 16 + l15) * 40], hi);
#pragma unroll
        for (int nt = 0; nt < 4; ++nt) {
            v16bf b = frag16(&sW[(wn + nt * 16 + l15) * 40], hi);
#pragma unroll
            for (int mt = 0; mt < MT; ++mt)
                acc[mt][nt] = __builtin_amdgcn_wmma_f32_16x16x32_bf16(
                    false, a[mt], false, b, (short)0, acc[mt][nt], false, false);
        }
    }

    float bs = 1.0f;
    if (mode == 2 && pw1 && pw2) bs = (*pw1) + (*pw2);

#pragma unroll
    for (int mt = 0; mt < MT; ++mt) {
#pragma unroll
        for (int nt = 0; nt < 4; ++nt) {
#pragma unroll
            for (int r = 0; r < 8; ++r) {
                int gm = m0 + wm + mt * 16 + r + 8 * hi;
                int gn = n0 + wn + nt * 16 + l15;
                float v = acc[mt][nt][r];
                if (mode == 0) {
                    outb[(size_t)gm * N + gn] = f2bf(v + bias[gn]);
                } else if (mode == 1) {
                    outb[(size_t)gm * N + gn] = f2bf(gelu_exact(v + bias[gn]));
                } else {
                    outf[(size_t)gm * N + gn] = v + bias[gn] * bs + res[(size_t)gm * N + gn];
                }
            }
        }
    }
}

// ---------------------------------------------------------------------------
// Flash attention, bf16 WMMA with f32 online softmax. DH=64, S=1024.
// Grid: (S/64, H, B). Block 128 thr = 4 waves; each wave owns 16 q rows.
// Buffers are [4096, 768] bf16; head h at column h*64.
// ---------------------------------------------------------------------------
__global__ __launch_bounds__(128) void attn_kernel(
    const unsigned short* __restrict__ Qb,
    const unsigned short* __restrict__ Kb,
    const unsigned short* __restrict__ Vb,
    unsigned short* __restrict__ Ctx) {
    __shared__ __align__(16) unsigned short sQ[64 * 72];     // [qrow][d]
    __shared__ __align__(16) unsigned short sK[32 * 72];     // [key][d]
    __shared__ __align__(16) unsigned short sVt[64 * 36];    // [d][key]
    __shared__ __align__(16) unsigned short sP[4 * 16 * 36]; // per-wave P bf16
    __shared__ float sS[4 * 16 * 33];                        // per-wave scores
    __shared__ float sAl[4 * 16];
    __shared__ float sL[4 * 16];

    const int t = threadIdx.x;
    const int lane = t & 31;
    const int wave = t >> 5;
    const int hi = lane >> 4;
    const int l15 = lane & 15;
    const int q0 = blockIdx.x * 64;
    const int co = blockIdx.y * HDIM;
    const int rb = blockIdx.z * SEQ;
    const float scale = 0.125f; // 1/sqrt(64)

    // stage Q block (64 x 64) as b128 chunks
#pragma unroll
    for (int i = 0; i < 4; ++i) {
        int c = i * 128 + t;
        int r = c >> 3, d8 = c & 7;
        copy_b128_g2l(&Qb[(size_t)(rb + q0 + r) * DMODEL + co + d8 * 8],
                      &sQ[r * 72 + d8 * 8]);
    }
    async_copy_wait();

    v8f acc[4] = {};
    float mrow = -3.0e38f, lrow = 0.0f;

    for (int j = 0; j < SEQ; j += 32) {
        __syncthreads();
        // K chunk (32 x 64): contiguous b128 copies (async path if available)
#pragma unroll
        for (int i = 0; i < 2; ++i) {
            int c = i * 128 + t;
            int kr = c >> 3, d8 = c & 7;
            copy_b128_g2l(&Kb[(size_t)(rb + j + kr) * DMODEL + co + d8 * 8],
                          &sK[kr * 72 + d8 * 8]);
        }
        // V chunk transposed: vector global load + 8-way LDS scatter
#pragma unroll
        for (int i = 0; i < 2; ++i) {
            int c = i * 128 + t;
            int kr = c >> 3, d8 = c & 7;
            uint4 q = *(const uint4*)&Vb[(size_t)(rb + j + kr) * DMODEL + co + d8 * 8];
#pragma unroll
            for (int jj = 0; jj < 8; ++jj)
                sVt[(d8 * 8 + jj) * 36 + kr] = u16_of(q, jj);
        }
        async_copy_wait();
        __syncthreads();

        // S(16x32) = Q(16x64) @ K^T
#pragma unroll
        for (int nt = 0; nt < 2; ++nt) {
            v8f s = {};
#pragma unroll
            for (int kc = 0; kc < 2; ++kc) {
                v16bf a = frag16(&sQ[(wave * 16 + l15) * 72 + kc * 32], hi);
                v16bf b = frag16(&sK[(nt * 16 + l15) * 72 + kc * 32], hi);
                s = __builtin_amdgcn_wmma_f32_16x16x32_bf16(
                    false, a, false, b, (short)0, s, false, false);
            }
#pragma unroll
            for (int r = 0; r < 8; ++r)
                sS[wave * (16 * 33) + (r + 8 * hi) * 33 + nt * 16 + l15] = s[r];
        }
        __syncthreads();

        // online softmax, full wave: lanes L and L^16 split the 32 columns of
        // row (L&15) and combine via shfl_xor -> no EXEC divergence.
        {
            const float* srow = &sS[wave * (16 * 33) + l15 * 33 + hi * 16];
            float mx = -3.0e38f;
#pragma unroll
            for (int c = 0; c < 16; ++c) mx = fmaxf(mx, srow[c] * scale);
            mx = fmaxf(mx, __shfl_xor(mx, 16, 32));
            float nm = fmaxf(mrow, mx);
            float alpha = __expf(mrow - nm);
            float psum = 0.f;
            unsigned short* prow = &sP[wave * (16 * 36) + l15 * 36 + hi * 16];
#pragma unroll
            for (int c = 0; c < 16; ++c) {
                float p = __expf(srow[c] * scale - nm);
                psum += p;
                prow[c] = f2bf(p);
            }
            psum += __shfl_xor(psum, 16, 32);
            lrow = lrow * alpha + psum;
            mrow = nm;
            sAl[wave * 16 + l15] = alpha; // both halves write identical value
        }
        __syncthreads();

        // rescale accumulators, then ctx += P(16x32) @ V(32x64)
        v16bf ap = frag16(&sP[wave * (16 * 36) + l15 * 36], hi);
#pragma unroll
        for (int nt = 0; nt < 4; ++nt) {
#pragma unroll
            for (int r = 0; r < 8; ++r)
                acc[nt][r] *= sAl[wave * 16 + r + 8 * hi];
            v16bf bv = frag16(&sVt[(nt * 16 + l15) * 36], hi);
            acc[nt] = __builtin_amdgcn_wmma_f32_16x16x32_bf16(
                false, ap, false, bv, (short)0, acc[nt], false, false);
        }
    }

    sL[wave * 16 + l15] = lrow; // both halves identical
    __syncthreads();

#pragma unroll
    for (int nt = 0; nt < 4; ++nt) {
#pragma unroll
        for (int r = 0; r < 8; ++r) {
            int row = r + 8 * hi;
            float inv = 1.0f / sL[wave * 16 + row];
            int gRow = rb + q0 + wave * 16 + row;
            Ctx[(size_t)gRow * DMODEL + co + nt * 16 + l15] = f2bf(acc[nt][r] * inv);
        }
    }
}

// ---------------------------------------------------------------------------
// Host launch
// ---------------------------------------------------------------------------
extern "C" void kernel_launch(void* const* d_in, const int* in_sizes, int n_in,
                              void* d_out, int out_size, void* d_ws, size_t ws_size,
                              hipStream_t stream) {
    (void)in_sizes; (void)n_in; (void)out_size; (void)ws_size;
    const float* x   = (const float*)d_in[0];
    const float* y   = (const float*)d_in[1];
    const float* Wq  = (const float*)d_in[2];  const float* bq  = (const float*)d_in[3];
    const float* Wk  = (const float*)d_in[4];  const float* bk  = (const float*)d_in[5];
    const float* Wv  = (const float*)d_in[6];  const float* bv  = (const float*)d_in[7];
    const float* Wo  = (const float*)d_in[8];  const float* bo  = (const float*)d_in[9];
    const float* Wqd = (const float*)d_in[10]; const float* bqd = (const float*)d_in[11];
    const float* Wkd = (const float*)d_in[12]; const float* bkd = (const float*)d_in[13];
    const float* Wvd = (const float*)d_in[14]; const float* bvd = (const float*)d_in[15];
    const float* Wod = (const float*)d_in[16]; const float* bod = (const float*)d_in[17];
    const float* fc1_w  = (const float*)d_in[18]; const float* fc1_b  = (const float*)d_in[19];
    const float* fc2_w  = (const float*)d_in[20]; const float* fc2_b  = (const float*)d_in[21];
    const float* fc1d_w = (const float*)d_in[22]; const float* fc1d_b = (const float*)d_in[23];
    const float* fc2d_w = (const float*)d_in[24]; const float* fc2d_b = (const float*)d_in[25];
    const float* ln_a_g  = (const float*)d_in[26]; const float* ln_ad_g = (const float*)d_in[27];
    const float* ln_f_g  = (const float*)d_in[28]; const float* ln_fd_g = (const float*)d_in[29];
    const float* ln_a_b  = (const float*)d_in[30]; const float* ln_ad_b = (const float*)d_in[31];
    const float* ln_f_b  = (const float*)d_in[32]; const float* ln_fd_b = (const float*)d_in[33];
    const float* w11 = (const float*)d_in[34]; const float* w12 = (const float*)d_in[35];
    const float* w21 = (const float*)d_in[36]; const float* w22 = (const float*)d_in[37];

    char* ws = (char*)d_ws;
    size_t off = 0;
    auto alloc = [&](size_t bytes) -> void* {
        off = (off + 255) & ~(size_t)255;
        void* p = ws + off;
        off += bytes;
        return p;
    };
    typedef unsigned short u16;
    const size_t DD = (size_t)DMODEL * DMODEL, DF = (size_t)DMODEL * FDIM;
    const size_t MD = (size_t)MROWS * DMODEL, MF = (size_t)MROWS * FDIM;

    u16* wqB  = (u16*)alloc(DD * 2); u16* wkB  = (u16*)alloc(DD * 2);
    u16* wvB  = (u16*)alloc(DD * 2); u16* woB  = (u16*)alloc(DD * 2);
    u16* wqdB = (u16*)alloc(DD * 2); u16* wkdB = (u16*)alloc(DD * 2);
    u16* wvdB = (u16*)alloc(DD * 2); u16* wodB = (u16*)alloc(DD * 2);
    u16* fc1B  = (u16*)alloc(DF * 2); u16* fc2B  = (u16*)alloc(DF * 2);
    u16* fc1dB = (u16*)alloc(DF * 2); u16* fc2dB = (u16*)alloc(DF * 2);
    u16* Xn  = (u16*)alloc(MD * 2); u16* Yn  = (u16*)alloc(MD * 2);
    u16* Q   = (u16*)alloc(MD * 2); u16* Kk  = (u16*)alloc(MD * 2); u16* V   = (u16*)alloc(MD * 2);
    u16* Qd  = (u16*)alloc(MD * 2); u16* Kd  = (u16*)alloc(MD * 2); u16* Vd  = (u16*)alloc(MD * 2);
    u16* Csx = (u16*)alloc(MD * 2); u16* Csy = (u16*)alloc(MD * 2);
    u16* Ccx = (u16*)alloc(MD * 2); u16* Ccy = (u16*)alloc(MD * 2);
    u16* Xn2 = (u16*)alloc(MD * 2); u16* Yn2 = (u16*)alloc(MD * 2);
    u16* H1x = (u16*)alloc(MF * 2); u16* H1y = (u16*)alloc(MF * 2);
    float* X1 = (float*)alloc(MD * 4); float* Y1 = (float*)alloc(MD * 4);

    auto cvt = [&](const float* s, u16* d, size_t n) {
        int nchunk = (int)(n / 8);
        int grid = (nchunk + 255) / 256; if (grid > 2048) grid = 2048;
        cvt_f32_bf16<<<grid, 256, 0, stream>>>(s, d, nchunk);
    };
    cvt(Wq, wqB, DD);   cvt(Wk, wkB, DD);   cvt(Wv, wvB, DD);   cvt(Wo, woB, DD);
    cvt(Wqd, wqdB, DD); cvt(Wkd, wkdB, DD); cvt(Wvd, wvdB, DD); cvt(Wod, wodB, DD);
    cvt(fc1_w, fc1B, DF); cvt(fc2_w, fc2B, DF); cvt(fc1d_w, fc1dB, DF); cvt(fc2d_w, fc2dB, DF);

    // LN 1
    ln_bf16<<<MROWS, 256, 0, stream>>>(x, ln_a_g, ln_a_b, Xn);
    ln_bf16<<<MROWS, 256, 0, stream>>>(y, ln_ad_g, ln_ad_b, Yn);

    auto gemm1 = [&](const u16* A, int K, const u16* W, int N, const float* bias,
                     const float* res, u16* outb, float* outf, int mode,
                     const float* pw1, const float* pw2) {
        dim3 g(N / 128, MROWS / 64);
        gemm_bf16<1><<<g, 256, 0, stream>>>(A, K, W, N, bias, res, outb, outf,
                                            N, K, mode, pw1, pw2);
    };
    auto gemm2 = [&](const u16* A, int K, const u16* W, int N, const float* bias,
                     const float* res, u16* outb, float* outf, int mode,
                     const float* pw1, const float* pw2) {
        dim3 g(N / 128, MROWS / 128);
        gemm_bf16<2><<<g, 256, 0, stream>>>(A, K, W, N, bias, res, outb, outf,
                                            N, K, mode, pw1, pw2);
    };

    // QKV projections (bf16 out)
    gemm1(Xn, DMODEL, wqB,  DMODEL, bq,  nullptr, Q,  nullptr, 0, nullptr, nullptr);
    gemm1(Xn, DMODEL, wkB,  DMODEL, bk,  nullptr, Kk, nullptr, 0, nullptr, nullptr);
    gemm1(Xn, DMODEL, wvB,  DMODEL, bv,  nullptr, V,  nullptr, 0, nullptr, nullptr);
    gemm1(Yn, DMODEL, wqdB, DMODEL, bqd, nullptr, Qd, nullptr, 0, nullptr, nullptr);
    gemm1(Yn, DMODEL, wkdB, DMODEL, bkd, nullptr, Kd, nullptr, 0, nullptr, nullptr);
    gemm1(Yn, DMODEL, wvdB, DMODEL, bvd, nullptr, Vd, nullptr, 0, nullptr, nullptr);

    // attention: self + cross streams
    dim3 ag(SEQ / 64, NHEAD, 4);
    attn_kernel<<<ag, 128, 0, stream>>>(Q,  Kk, V,  Csx);
    attn_kernel<<<ag, 128, 0, stream>>>(Qd, Kd, Vd, Csy);
    attn_kernel<<<ag, 128, 0, stream>>>(Q,  Kd, Vd, Ccx);
    attn_kernel<<<ag, 128, 0, stream>>>(Qd, Kk, V,  Ccy);

    // fold combine before the shared output projection:
    // w11*sx + w12*cx = (w11*ctx_s + w12*ctx_c) @ Wo + (w11+w12)*bo
    {
        int nchunk = (int)(MD / 8);
        int grid = (nchunk + 255) / 256; if (grid > 2048) grid = 2048;
        combine_bf16<<<grid, 256, 0, stream>>>(Csx, Ccx, w11, w12, Csx, nchunk);
        combine_bf16<<<grid, 256, 0, stream>>>(Csy, Ccy, w21, w22, Csy, nchunk);
    }

    // output projection + residual (f32)
    gemm1(Csx, DMODEL, woB,  DMODEL, bo,  x, nullptr, X1, 2, w11, w12);
    gemm1(Csy, DMODEL, wodB, DMODEL, bod, y, nullptr, Y1, 2, w21, w22);

    // LN 2
    ln_bf16<<<MROWS, 256, 0, stream>>>(X1, ln_f_g, ln_f_b, Xn2);
    ln_bf16<<<MROWS, 256, 0, stream>>>(Y1, ln_fd_g, ln_fd_b, Yn2);

    // MLP: gelu(xn @ fc1 + b) @ fc2 + b + residual
    gemm2(Xn2, DMODEL, fc1B,  FDIM, fc1_b,  nullptr, H1x, nullptr, 1, nullptr, nullptr);
    gemm2(Yn2, DMODEL, fc1dB, FDIM, fc1d_b, nullptr, H1y, nullptr, 1, nullptr, nullptr);

    float* outX = (float*)d_out;
    float* outY = outX + MD;
    gemm1(H1x, FDIM, fc2B,  DMODEL, fc2_b,  X1, nullptr, outX, 2, nullptr, nullptr);
    gemm1(H1y, FDIM, fc2dB, DMODEL, fc2d_b, Y1, nullptr, outY, 2, nullptr, nullptr);
}
